// SmallSTGCN_83631603188222
// MI455X (gfx1250) — matrix operand order repaired and checked by reading, not compiled, not run on hardware
//
#include <hip/hip_runtime.h>
#include <hip/hip_bf16.h>
#include <stdint.h>

typedef __bf16 bf16;
typedef __attribute__((ext_vector_type(8)))  bf16  v8bf;
typedef __attribute__((ext_vector_type(16))) bf16  v16bf;
typedef __attribute__((ext_vector_type(8)))  float v8f;

#define DEV static __device__ __forceinline__

// ---------------------------------------------------------------------------
// WMMA helpers (gfx1250, wave32). D = A(16xK) * B(Kx16) + C, K=32 bf16.
// A frag: lane L -> row M=L%16; halves e: K = e + 8*(e>=8) + 8*(L>=16)
//   => two contiguous 8-half runs at K0 = 8*(L>=16) and K0+16.
// B frag: lane L -> col N=L%16; halves e: K as for A (same table order).
// C/D frag: vgpr r, lane L -> (M = r + 8*(L>=16), N = L%16).
// ---------------------------------------------------------------------------
DEV v8f wmma_bf(v16bf a, v16bf b, v8f c) {
  return __builtin_amdgcn_wmma_f32_16x16x32_bf16(false, a, false, b, (short)0, c,
                                                 false, false);
}

DEV v8f vzero8() {
  v8f c;
#pragma unroll
  for (int i = 0; i < 8; ++i) c[i] = 0.0f;
  return c;
}

DEV v16bf combine16(v8bf lo, v8bf hi) {
  v16bf r;
#pragma unroll
  for (int i = 0; i < 8; ++i) { r[i] = lo[i]; r[i + 8] = hi[i]; }
  return r;
}

// rowbase points at the 32-element K-chunk for this lane's M row.
DEV v16bf afrag(const bf16* rowbase, int lane) {
  const int a = (lane & 16) ? 8 : 0;
  v8bf lo = *(const v8bf*)(rowbase + a);
  v8bf hi = *(const v8bf*)(rowbase + 16 + a);
  return combine16(lo, hi);
}

// W is [Krows][ldn] row-major; fragment covers K rows 0..31, col n0 + lane%16, K in A's table order
DEV v16bf bfrag(const bf16* W, int ldn, int n0, int lane) {
  const bf16* p = W + n0 + (lane & 15);
  const int a = (lane & 16) ? 8 : 0;
  v16bf r;
#pragma unroll
  for (int i = 0; i < 8; ++i) {
    r[i] = p[(size_t)(a + i) * ldn];
    r[i + 8] = p[(size_t)(16 + a + i) * ldn];
  }
  return r;
}

// Same but rows >= kmax supply zeros (for 30-node operands padded to K=32).
DEV v16bf bfrag_rows(const bf16* W, int ldn, int n0, int lane, int kmax) {
  const bf16* p = W + n0 + (lane & 15);
  const int a = (lane & 16) ? 8 : 0;
  v16bf r;
#pragma unroll
  for (int i = 0; i < 8; ++i) {
    r[i] = (a + i < kmax) ? p[(size_t)(a + i) * ldn] : (bf16)0.0f;
    r[i + 8] = (16 + a + i < kmax) ? p[(size_t)(16 + a + i) * ldn] : (bf16)0.0f;
  }
  return r;
}

// Fast sigmoid pieces: hardware v_rcp_f32 (~1 ulp) instead of IEEE divide.
DEV float fastrcp(float x) { return __builtin_amdgcn_rcpf(x); }
DEV float sigmoidf(float x) { return fastrcp(1.0f + __expf(-x)); }
DEV float gate_relu(float a, float b, float c) {
  // relu(a * sigmoid(b) + c)
  return fmaxf(fmaf(a, sigmoidf(b), c), 0.0f);
}

// Async global -> LDS copy of 16 bytes (CDNA5, ASYNCcnt-tracked).
DEV void async_ld16(void* ldsdst, const void* sbase, unsigned byteoff) {
  unsigned ldsoff = (unsigned)(uintptr_t)ldsdst;
  asm volatile("global_load_async_to_lds_b128 %0, %1, %2 offset:0"
               :
               : "v"(ldsoff), "v"(byteoff), "s"(sbase)
               : "memory");
}
DEV void async_wait0() {
  asm volatile("s_wait_asynccnt 0x0" ::: "memory");
}

// ---------------------------------------------------------------------------
// Constants / layouts
// ---------------------------------------------------------------------------
#define TBATCH 4    // samples per workgroup in the fused ST kernel
#define STW 16      // waves per WG in k_stgcn (512 threads)
#define MLW 16      // waves per WG in k_mlp (512 threads)

// LDS weight arena layout (bf16 element offsets)
#define OFF_S      0        // [32][32] graph operator, zero padded
#define OFF_TC1_1  1024     // st1.tc1: 3 x [32][32] (K=3*6 padded to 32)
#define OFF_CHEB1  4096     // st1.cheb: W0 [32][32], W1 [32][32]
#define OFF_TC2_1  6144     // st1.tc2: 3 x [96][32]
#define OFF_TC1_2  15360    // st2.tc1: 3 x [96][32]
#define OFF_CHEB2  24576    // st2.cheb: W0, W1
#define OFF_TC2_2  26624    // st2.tc2: 3 x [96][32]
#define NW_ELEMS   35840

// float bias arena (element offsets)
#define FB_TC1_1 0          // 3 x 32
#define FB_CHB1  96
#define FB_TC2_1 128        // 3 x 32
#define FB_BNS1  224        // gamma/sqrt(1+eps), padded to 32
#define FB_BNB1  256
#define FB_TC1_2 288
#define FB_CHB2  384
#define FB_TC2_2 416
#define FB_BNS2  512
#define FB_BNB2  544
#define NFB_ELEMS 576

// workspace byte offsets
#define WS_SW    0u
#define WS_FB    71680u                 // 35840 * 2
#define WS_FC1W  73984u                 // + 576*4 (=2304), 256B aligned
#define WS_FC2W  1057024u               // + 1920*256*2
#define WS_FEAT  1523968u               // + 256*912*2 ; feat: 8192*1920*2

// d_in leaf order (jax pytree: sorted dict keys)
enum {
  IX_X = 0, IX_EDGE = 1, IX_FC1B = 2, IX_FC1W = 3, IX_FC2B = 4, IX_FC2W = 5,
  IX_S1_BETA = 6, IX_S1_W0 = 7, IX_S1_W1 = 8, IX_S1_CHB = 9, IX_S1_GAMMA = 10,
  IX_S1_T1B1 = 11, IX_S1_T1B2 = 12, IX_S1_T1B3 = 13,
  IX_S1_T1W1 = 14, IX_S1_T1W2 = 15, IX_S1_T1W3 = 16,
  IX_S1_T2B1 = 17, IX_S1_T2B2 = 18, IX_S1_T2B3 = 19,
  IX_S1_T2W1 = 20, IX_S1_T2W2 = 21, IX_S1_T2W3 = 22,
  IX_S2_BETA = 23, IX_S2_W0 = 24, IX_S2_W1 = 25, IX_S2_CHB = 26, IX_S2_GAMMA = 27,
  IX_S2_T1B1 = 28, IX_S2_T1B2 = 29, IX_S2_T1B3 = 30,
  IX_S2_T1W1 = 31, IX_S2_T1W2 = 32, IX_S2_T1W3 = 33,
  IX_S2_T2B1 = 34, IX_S2_T2B2 = 35, IX_S2_T2B3 = 36,
  IX_S2_T2W1 = 37, IX_S2_T2W2 = 38, IX_S2_T2W3 = 39
};

struct PtrPack { const void* p[40]; };

// ---------------------------------------------------------------------------
// Prep kernel: build S from edge_index, repack small weights/biases to ws.
// ---------------------------------------------------------------------------
__global__ __launch_bounds__(1024) void k_prep(PtrPack pk, bf16* __restrict__ wsSW,
                                               float* __restrict__ wsFB) {
  const int tid = threadIdx.x;
  __shared__ float sA[900];
  __shared__ float sdinv[30];

  for (int i = tid; i < 900; i += 1024) sA[i] = 0.0f;
  __syncthreads();
  if (tid < 120) {
    const int* e = (const int*)pk.p[IX_EDGE];
    int r = e[tid], c = e[120 + tid];
    if (r != c) atomicAdd(&sA[r * 30 + c], 1.0f);
  }
  __syncthreads();
  if (tid < 30) {
    float d = 0.0f;
    for (int m = 0; m < 30; ++m) d += sA[tid * 30 + m];
    sdinv[tid] = (d > 0.0f) ? rsqrtf(fmaxf(d, 1e-12f)) : 0.0f;
  }
  __syncthreads();
  {  // S = -(dinv A dinv), padded [32][32]
    int n = tid >> 5, m = tid & 31;
    float v = (n < 30 && m < 30) ? -(sdinv[n] * sA[n * 30 + m] * sdinv[m]) : 0.0f;
    wsSW[OFF_S + tid] = (bf16)v;
  }

  // st1.tc1 weights: (32,6,1,3) -> [K=kk*6+ci pad 32][co]
  for (int j = 0; j < 3; ++j) {
    const float* w = (const float*)pk.p[IX_S1_T1W1 + j];
    bf16* dst = wsSW + OFF_TC1_1 + j * 1024;
    for (int i = tid; i < 1024; i += 1024) {
      int Kidx = i >> 5, co = i & 31;
      float v = 0.0f;
      if (Kidx < 18) { int kk = Kidx / 6, ci = Kidx - kk * 6; v = w[(co * 6 + ci) * 3 + kk]; }
      dst[i] = (bf16)v;
    }
  }
  // cin=32 convs: (32,32,1,3) -> [K=kk*32+ci][co]
  auto pack96 = [&](const float* w, bf16* dst) {
    for (int i = tid; i < 96 * 32; i += 1024) {
      int Kidx = i >> 5, co = i & 31;
      int kk = Kidx >> 5, ci = Kidx & 31;
      dst[i] = (bf16)w[(co * 32 + ci) * 3 + kk];
    }
  };
  for (int j = 0; j < 3; ++j) pack96((const float*)pk.p[IX_S1_T2W1 + j], wsSW + OFF_TC2_1 + j * 3072);
  for (int j = 0; j < 3; ++j) pack96((const float*)pk.p[IX_S2_T1W1 + j], wsSW + OFF_TC1_2 + j * 3072);
  for (int j = 0; j < 3; ++j) pack96((const float*)pk.p[IX_S2_T2W1 + j], wsSW + OFF_TC2_2 + j * 3072);

  auto cpy1024 = [&](const float* w, bf16* dst) {
    for (int i = tid; i < 1024; i += 1024) dst[i] = (bf16)w[i];
  };
  cpy1024((const float*)pk.p[IX_S1_W0], wsSW + OFF_CHEB1);
  cpy1024((const float*)pk.p[IX_S1_W1], wsSW + OFF_CHEB1 + 1024);
  cpy1024((const float*)pk.p[IX_S2_W0], wsSW + OFF_CHEB2);
  cpy1024((const float*)pk.p[IX_S2_W1], wsSW + OFF_CHEB2 + 1024);

  // biases and BN params
  const int bsrc[14] = {IX_S1_T1B1, IX_S1_T1B2, IX_S1_T1B3, IX_S1_CHB,
                        IX_S1_T2B1, IX_S1_T2B2, IX_S1_T2B3,
                        IX_S2_T1B1, IX_S2_T1B2, IX_S2_T1B3, IX_S2_CHB,
                        IX_S2_T2B1, IX_S2_T2B2, IX_S2_T2B3};
  const int bdst[14] = {FB_TC1_1, FB_TC1_1 + 32, FB_TC1_1 + 64, FB_CHB1,
                        FB_TC2_1, FB_TC2_1 + 32, FB_TC2_1 + 64,
                        FB_TC1_2, FB_TC1_2 + 32, FB_TC1_2 + 64, FB_CHB2,
                        FB_TC2_2, FB_TC2_2 + 32, FB_TC2_2 + 64};
  if (tid < 32) {
    for (int e2 = 0; e2 < 14; ++e2)
      wsFB[bdst[e2] + tid] = ((const float*)pk.p[bsrc[e2]])[tid];
    float inv = rsqrtf(1.0f + 1e-5f);
    wsFB[FB_BNS1 + tid] = (tid < 30) ? ((const float*)pk.p[IX_S1_GAMMA])[tid] * inv : 0.0f;
    wsFB[FB_BNB1 + tid] = (tid < 30) ? ((const float*)pk.p[IX_S1_BETA])[tid] : 0.0f;
    wsFB[FB_BNS2 + tid] = (tid < 30) ? ((const float*)pk.p[IX_S2_GAMMA])[tid] * inv : 0.0f;
    wsFB[FB_BNB2 + tid] = (tid < 30) ? ((const float*)pk.p[IX_S2_BETA])[tid] : 0.0f;
  }
}

// fc weights -> bf16 (fc2 zero-padded to N=912)
__global__ __launch_bounds__(256) void k_fcconv(const float* __restrict__ fc1w,
                                                const float* __restrict__ fc2w,
                                                bf16* __restrict__ d1,
                                                bf16* __restrict__ d2) {
  const int stride = gridDim.x * blockDim.x;
  const int t0 = blockIdx.x * blockDim.x + threadIdx.x;
  for (int i = t0; i < 1920 * 256; i += stride) d1[i] = (bf16)fc1w[i];
  for (int i = t0; i < 256 * 912; i += stride) {
    int row = i / 912, c = i - row * 912;
    d2[i] = (c < 900) ? (bf16)fc2w[row * 900 + c] : (bf16)0.0f;
  }
}

// ---------------------------------------------------------------------------
// Fused ST-block kernel: per WG processes TBATCH samples entirely in LDS.
// Activation layout [b][T][30][32] is fully contiguous, so the output address
// of GEMM row m is simply base + m*32 + col (no (b,t,n) decode needed).
// ---------------------------------------------------------------------------
template <int Tin, int Tout, int K, bool PACKED, bool BN, bool FINAL>
DEV void tconv_stage(const bf16* __restrict__ actin, bf16* __restrict__ actout,
                     const bf16* __restrict__ W, const float* __restrict__ fb,
                     const float* __restrict__ bnsb, bf16* __restrict__ featg,
                     int b0, int lane, int wave) {
  constexpr int ROWS = TBATCH * Tout * 30;
  constexpr int NTILE = (ROWS / 16) * 2;
  constexpr int KS = K / 32;
  const int lrow = lane & 15;
  const int hi8 = (lane & 16) ? 8 : 0;
  for (int tile = wave; tile < NTILE; tile += STW) {
    const int mt = tile >> 1, nt = tile & 1;
    const int n0 = nt * 16;
    const int row = mt * 16 + lrow;                 // this lane's A row
    const int bo = row / (Tout * 30);               // sample of A row
    // input window base: flat(b,t+ks,n) = row + bo*(Tin-Tout)*30 + ks*30
    const int inoff = row + bo * (Tin - Tout) * 30;
    v8f c1 = vzero8(), c2 = vzero8(), c3 = vzero8();
#pragma unroll
    for (int ks = 0; ks < KS; ++ks) {
      const bf16* abase = PACKED ? (actin + row * 32)
                                 : (actin + ((inoff + ks * 30) << 5));
      v16bf va = afrag(abase, lane);
      c1 = wmma_bf(va, bfrag(W + (0 * K + ks * 32) * 32, 32, n0, lane), c1);
      c2 = wmma_bf(va, bfrag(W + (1 * K + ks * 32) * 32, 32, n0, lane), c2);
      c3 = wmma_bf(va, bfrag(W + (2 * K + ks * 32) * 32, 32, n0, lane), c3);
    }
    const int col = n0 + lrow;
    const float bb1 = fb[col], bb2 = fb[32 + col], bb3 = fb[64 + col];
    const int m0 = mt * 16 + hi8;                   // first D row of this half
    int no = m0 % 30;                               // node id (for BN), wraps
    bf16* outp = FINAL ? nullptr : (actout + m0 * 32 + col);
    bf16* fout = FINAL ? (featg + (size_t)b0 * 1920 + m0 * 32 + col) : nullptr;
#pragma unroll
    for (int r = 0; r < 8; ++r) {
      float o = gate_relu(c1[r] + bb1, c2[r] + bb2, c3[r] + bb3);
      if constexpr (BN) {
        o = fmaf(o, bnsb[no], bnsb[32 + no]);
        o = fmaxf(o, 0.0f);
      }
      bf16 v = (bf16)o;
      if constexpr (FINAL) fout[r * 32] = v;
      else outp[r * 32] = v;
      ++no;
      if (no == 30) no = 0;
    }
  }
}

// ChebConv K=2 + relu, in place: X <- relu(X W0 + (S X) W1 + b)
template <int T>
DEV void cheb_stage(bf16* __restrict__ act, const bf16* __restrict__ S,
                    const bf16* __restrict__ W0, const bf16* __restrict__ W1,
                    const float* __restrict__ cb, bf16* __restrict__ ys,
                    int lane, int wave) {
  const int lrow = lane & 15;
  const int hi8 = (lane & 16) ? 8 : 0;
  constexpr int PAIRS = TBATCH * T;
  for (int p = wave; p < PAIRS; p += STW) {
    bf16* X = act + (size_t)p * 30 * 32;
    // Y = S @ X -> ys[32][32]
#pragma unroll
    for (int mt = 0; mt < 2; ++mt) {
      v16bf sa = afrag(S + (mt * 16 + lrow) * 32, lane);
#pragma unroll
      for (int nt = 0; nt < 2; ++nt) {
        v8f c = wmma_bf(sa, bfrag_rows(X, 32, nt * 16, lane, 30), vzero8());
        const int col = nt * 16 + lrow;
        bf16* yp = ys + (mt * 16 + hi8) * 32 + col;
#pragma unroll
        for (int r = 0; r < 8; ++r) yp[r * 32] = (bf16)c[r];
      }
    }
    // Out = X W0 + Y W1 (+bias, relu), in place (reads staged before stores)
    v16bf xa[2], ya[2];
#pragma unroll
    for (int mt = 0; mt < 2; ++mt) {
      xa[mt] = afrag(X + (mt * 16 + lrow) * 32, lane);
      ya[mt] = afrag(ys + (mt * 16 + lrow) * 32, lane);
    }
#pragma unroll
    for (int mt = 0; mt < 2; ++mt) {
#pragma unroll
      for (int nt = 0; nt < 2; ++nt) {
        v8f c = wmma_bf(xa[mt], bfrag(W0, 32, nt * 16, lane), vzero8());
        c = wmma_bf(ya[mt], bfrag(W1, 32, nt * 16, lane), c);
        const int col = nt * 16 + lrow;
        const float bb = cb[col];
#pragma unroll
        for (int r = 0; r < 8; ++r) {
          const int m = mt * 16 + r + hi8;
          if (m < 30) X[m * 32 + col] = (bf16)fmaxf(c[r] + bb, 0.0f);
        }
      }
    }
  }
}

__global__ __launch_bounds__(512) void k_stgcn(const float* __restrict__ x,
                                               const bf16* __restrict__ wsSW,
                                               const float* __restrict__ wsFB,
                                               bf16* __restrict__ feat) {
  __shared__ __align__(16) bf16 sW[NW_ELEMS];
  __shared__ __align__(16) float sFB[NFB_ELEMS];
  __shared__ __align__(16) bf16 bufA[TBATCH * 8 * 30 * 32 + 128];  // pack / act3
  __shared__ __align__(16) bf16 bufB[TBATCH * 8 * 30 * 32 + 128];  // act1
  __shared__ __align__(16) bf16 bufC[TBATCH * 6 * 30 * 32 + 128];  // act2
  __shared__ __align__(16) bf16 sYS[STW * 1024];                   // per-wave cheb scratch

  const int tid = threadIdx.x;
  const int lane = tid & 31, wave = tid >> 5;
  const int b0 = blockIdx.x * TBATCH;

  // Async DMA the weight/bias arenas into LDS (overlaps with x repacking below)
  for (int i = tid; i < NW_ELEMS / 8; i += 512)
    async_ld16(sW + i * 8, wsSW, (unsigned)(i * 16));
  for (int i = tid; i < NFB_ELEMS / 4; i += 512)
    async_ld16(sFB + i * 4, wsFB, (unsigned)(i * 16));

  // Pack stage-1 GEMM rows: K = [x(t),x(t+1),x(t+2)] (6 ch each), padded to 32
  for (int idx = tid; idx < TBATCH * 8 * 30; idx += 512) {
    int b = idx / 240, rem = idx - b * 240;
    int t = rem / 30, n = rem - t * 30;
    const float* xp = x + ((size_t)(b0 + b) * 10 + t) * 180 + n * 6;
    bf16* dst = bufA + idx * 32;
#pragma unroll
    for (int kk = 0; kk < 3; ++kk)
#pragma unroll
      for (int ci = 0; ci < 6; ++ci) dst[kk * 6 + ci] = (bf16)xp[kk * 180 + ci];
#pragma unroll
    for (int j = 18; j < 32; ++j) dst[j] = (bf16)0.0f;
  }
  async_wait0();
  __syncthreads();

  tconv_stage<10, 8, 32, true, false, false>(bufA, bufB, sW + OFF_TC1_1,
                                             sFB + FB_TC1_1, nullptr, nullptr, b0, lane, wave);
  __syncthreads();
  cheb_stage<8>(bufB, sW + OFF_S, sW + OFF_CHEB1, sW + OFF_CHEB1 + 1024,
                sFB + FB_CHB1, sYS + wave * 1024, lane, wave);
  __syncthreads();
  tconv_stage<8, 6, 96, false, true, false>(bufB, bufC, sW + OFF_TC2_1,
                                            sFB + FB_TC2_1, sFB + FB_BNS1, nullptr, b0, lane, wave);
  __syncthreads();
  tconv_stage<6, 4, 96, false, false, false>(bufC, bufA, sW + OFF_TC1_2,
                                             sFB + FB_TC1_2, nullptr, nullptr, b0, lane, wave);
  __syncthreads();
  cheb_stage<4>(bufA, sW + OFF_S, sW + OFF_CHEB2, sW + OFF_CHEB2 + 1024,
                sFB + FB_CHB2, sYS + wave * 1024, lane, wave);
  __syncthreads();
  tconv_stage<4, 2, 96, false, true, true>(bufA, nullptr, sW + OFF_TC2_2,
                                           sFB + FB_TC2_2, sFB + FB_BNS2, feat, b0, lane, wave);
}

// ---------------------------------------------------------------------------
// MLP head: [B,1920] -> relu fc1 (256) -> fc2 (900) -> sigmoid.
// M-tile = 128 rows/WG (512 threads, 16 waves); each wave owns a 16-wide
// N slice of fc1 output, so fc1_w streams from L2 exactly once per WG.
// Feature K-chunks staged to LDS with async-to-LDS DMA.
// ---------------------------------------------------------------------------
__global__ __launch_bounds__(512) void k_mlp(const bf16* __restrict__ feat,
                                             const bf16* __restrict__ w1,
                                             const bf16* __restrict__ w2,
                                             const float* __restrict__ b1,
                                             const float* __restrict__ b2,
                                             float* __restrict__ out) {
  __shared__ __align__(16) bf16 sFeat[128 * 320];
  __shared__ __align__(16) bf16 sHid[128 * 256];
  const int tid = threadIdx.x, lane = tid & 31, wave = tid >> 5;
  const int lrow = lane & 15;
  const int hi8 = (lane & 16) ? 8 : 0;
  const int b0 = blockIdx.x * 128;
  const int wN0 = wave * 16;  // 16 waves x 16 cols = 256

  v8f acc[8];
#pragma unroll
  for (int i = 0; i < 8; ++i) acc[i] = vzero8();

  for (int kc = 0; kc < 6; ++kc) {  // K = 6 chunks of 320
    for (int i = tid; i < 128 * 40; i += 512) {
      int rrow = i / 40, c8 = i - rrow * 40;
      unsigned goff = ((unsigned)(b0 + rrow) * 1920u + (unsigned)kc * 320u + (unsigned)c8 * 8u) * 2u;
      async_ld16(sFeat + rrow * 320 + c8 * 8, feat, goff);
    }
    async_wait0();
    __syncthreads();
#pragma unroll 1
    for (int ks = 0; ks < 10; ++ks) {
      const int krow = kc * 320 + ks * 32;
      v16bf vb = bfrag(w1 + (size_t)krow * 256, 256, wN0, lane);
#pragma unroll
      for (int m8 = 0; m8 < 8; ++m8) {
        v16bf va = afrag(sFeat + (m8 * 16 + lrow) * 320 + ks * 32, lane);
        acc[m8] = wmma_bf(va, vb, acc[m8]);
      }
    }
    __syncthreads();
  }
  // bias + relu -> hidden (bf16)
  {
    const int col = wN0 + lrow;
    const float bb = b1[col];
#pragma unroll
    for (int m8 = 0; m8 < 8; ++m8) {
      bf16* hp = sHid + (m8 * 16 + hi8) * 256 + col;
#pragma unroll
      for (int r = 0; r < 8; ++r)
        hp[r * 256] = (bf16)fmaxf(acc[m8][r] + bb, 0.0f);
    }
  }
  __syncthreads();
  // fc2 (N padded to 912) + sigmoid
  for (int ntile = wave; ntile < 57; ntile += MLW) {
    const int n0 = ntile * 16;
    v16bf vb[8];
#pragma unroll
    for (int ks = 0; ks < 8; ++ks)
      vb[ks] = bfrag(w2 + (size_t)(ks * 32) * 912, 912, n0, lane);
    const int col = n0 + lrow;
    const float bb = (col < 900) ? b2[col] : 0.0f;
#pragma unroll 1
    for (int m8 = 0; m8 < 8; ++m8) {
      v8f c = vzero8();
#pragma unroll
      for (int ks = 0; ks < 8; ++ks) {
        v16bf va = afrag(sHid + (m8 * 16 + lrow) * 256 + ks * 32, lane);
        c = wmma_bf(va, vb[ks], c);
      }
      if (col < 900) {
#pragma unroll
        for (int r = 0; r < 8; ++r) {
          const int rrow = b0 + m8 * 16 + r + hi8;
          out[(size_t)rrow * 900 + col] = sigmoidf(c[r] + bb);
        }
      }
    }
  }
}

// ---------------------------------------------------------------------------
extern "C" void kernel_launch(void* const* d_in, const int* in_sizes, int n_in,
                              void* d_out, int out_size, void* d_ws, size_t ws_size,
                              hipStream_t stream) {
  (void)in_sizes; (void)out_size; (void)ws_size;
  PtrPack pk{};
  for (int i = 0; i < n_in && i < 40; ++i) pk.p[i] = d_in[i];

  char* ws = (char*)d_ws;
  bf16*  wsSW   = (bf16*)(ws + WS_SW);
  float* wsFB   = (float*)(ws + WS_FB);
  bf16*  wsFC1  = (bf16*)(ws + WS_FC1W);
  bf16*  wsFC2  = (bf16*)(ws + WS_FC2W);
  bf16*  wsFeat = (bf16*)(ws + WS_FEAT);

  k_prep<<<1, 1024, 0, stream>>>(pk, wsSW, wsFB);
  k_fcconv<<<512, 256, 0, stream>>>((const float*)d_in[IX_FC1W],
                                    (const float*)d_in[IX_FC2W], wsFC1, wsFC2);
  k_stgcn<<<8192 / TBATCH, 512, 0, stream>>>((const float*)d_in[IX_X], wsSW, wsFB, wsFeat);
  k_mlp<<<8192 / 128, 512, 0, stream>>>(wsFeat, wsFC1, wsFC2,
                                        (const float*)d_in[IX_FC1B],
                                        (const float*)d_in[IX_FC2B], (float*)d_out);
}
